// E3Transformer_38225208934662
// MI455X (gfx1250) — matrix operand order repaired and checked by reading, not compiled, not run on hardware
//
#include <hip/hip_runtime.h>
#include <hip/hip_bf16.h>
#include <math.h>

// ---------------------------------------------------------------------------
// CDNA5 (gfx1250) WMMA helpers: v_wmma_f32_16x16x32_bf16, wave32.
// ---------------------------------------------------------------------------
typedef __attribute__((ext_vector_type(16))) __bf16 v16bf;
typedef __attribute__((ext_vector_type(8)))  float  v8f;

#define E_EDGES 320000
#define N_NODES_C 10000
#define N_GRAPHS_C 64

__device__ __forceinline__ v8f wmma_bf(v16bf a, v16bf b, v8f c) {
  return __builtin_amdgcn_wmma_f32_16x16x32_bf16(false, a, false, b, (short)0, c,
                                                 false, false);
}

__device__ __forceinline__ v8f zero8() {
  v8f z;
#pragma unroll
  for (int j = 0; j < 8; ++j) z[j] = 0.f;
  return z;
}
__device__ __forceinline__ v8f splat8(float b) {
  v8f z;
#pragma unroll
  for (int j = 0; j < 8; ++j) z[j] = b;
  return z;
}

// A-matrix 16x32 bf16 K-index map (ISA 7.12.2): lanes 0-15 M=row K pairs
// {0,1},{2,3},{4,5},{6,7},{16,17},{18,19},{20,21},{22,23}; lanes 16-31 +8.
__device__ __forceinline__ int a_kmap(int p, int hi) {
  return ((p < 4) ? 2 * p : 2 * p + 8) + hi * 8;
}
// B-matrix 32x16: lane = column; same K map as A (lanes 0-15 K 0-7,16-23; lanes 16-31 +8).
__device__ __forceinline__ int b_kmap(int p, int hi) { return ((p < 4) ? 2 * p : 2 * p + 8) + hi * 8; }

// Pack A tile from a per-lane row pointer of bf16 (row-major, K=32 full).
__device__ __forceinline__ v16bf packA_bf16row(const __bf16* rowp, int hi) {
  v16bf a;
#pragma unroll
  for (int p = 0; p < 8; ++p) {
    int k = a_kmap(p, hi);
    a[2 * p] = rowp[k];
    a[2 * p + 1] = rowp[k + 1];
  }
  return a;
}
// Pack A tile from LDS bf16 row-major [16][ldm], k window at kBase.
__device__ __forceinline__ v16bf packA_lds(const __bf16* base, int ldm, int kBase) {
  int lane = threadIdx.x & 31;
  int row = lane & 15, hi = lane >> 4;
  v16bf a;
#pragma unroll
  for (int p = 0; p < 8; ++p) {
    int k = kBase + a_kmap(p, hi);
    a[2 * p] = base[row * ldm + k];
    a[2 * p + 1] = base[row * ldm + k + 1];
  }
  return a;
}
// Pack A tile from per-lane f32 row pointer with element stride, zero-padded K.
__device__ __forceinline__ v16bf packA_f32row(const float* rowp, int kstride,
                                              int kBase, int kValid) {
  int hi = (threadIdx.x >> 4) & 1;
  v16bf a;
#pragma unroll
  for (int p = 0; p < 8; ++p) {
    int k = kBase + a_kmap(p, hi);
    a[2 * p]     = (k < kValid)     ? (__bf16)rowp[k * kstride]       : (__bf16)0.f;
    a[2 * p + 1] = (k + 1 < kValid) ? (__bf16)rowp[(k + 1) * kstride] : (__bf16)0.f;
  }
  return a;
}
// Pack B tile from LDS weights stored column-major [col][Kpad] bf16.
__device__ __forceinline__ v16bf packB(const __bf16* wcm, int Kpad, int colBase,
                                       int kBase, int colsValid) {
  int lane = threadIdx.x & 31;
  int col = lane & 15, hi = lane >> 4;
  bool ok = col < colsValid;
  const __bf16* cp = wcm + (colBase + col) * Kpad;
  v16bf b;
#pragma unroll
  for (int p = 0; p < 8; ++p) {
    int k = kBase + b_kmap(p, hi);
    b[2 * p]     = ok ? cp[k]     : (__bf16)0.f;
    b[2 * p + 1] = ok ? cp[k + 1] : (__bf16)0.f;
  }
  return b;
}
// C/D layout: lane holds col = lane&15, rows M = v + 8*(lane>>4), v = 0..7.
__device__ __forceinline__ void storeC_ldsbf(__bf16* base, int ldm, int colBase, v8f c) {
  int lane = threadIdx.x & 31;
  int col = colBase + (lane & 15), hi = lane >> 4;
#pragma unroll
  for (int v = 0; v < 8; ++v) base[(v + 8 * hi) * ldm + col] = (__bf16)c[v];
}
__device__ __forceinline__ void storeC_ldsf(float* base, int ldm, int colBase, v8f c) {
  int lane = threadIdx.x & 31;
  int col = colBase + (lane & 15), hi = lane >> 4;
#pragma unroll
  for (int v = 0; v < 8; ++v) base[(v + 8 * hi) * ldm + col] = c[v];
}

// Monotonic float<->int key for atomicMax-based segment max.
__device__ __forceinline__ int fenc(float f) {
  int i = __float_as_int(f);
  return i >= 0 ? i : (i ^ 0x7fffffff);
}
__device__ __forceinline__ float fdec(int k) {
  return __int_as_float(k >= 0 ? k : (k ^ 0x7fffffff));
}
__device__ __forceinline__ float silu(float x) { return x / (1.f + __expf(-x)); }

// ---------------------------------------------------------------------------
// Kernel: per-edge geometry -> Y1 (E x 3), Y2 (E x 5), edge_attr bf16 (E x 32)
// ---------------------------------------------------------------------------
__global__ void geom_kernel(const float* __restrict__ pos, const int* __restrict__ srcI,
                            const int* __restrict__ dstI, const int* __restrict__ bond,
                            const float* __restrict__ ebond, __bf16* __restrict__ attr,
                            float* __restrict__ Y1g, float* __restrict__ Y2g, int E) {
  int e = blockIdx.x * blockDim.x + threadIdx.x;
  if (e >= E) return;
  int s = srcI[e], d = dstI[e];
  float ex = pos[s * 3 + 0] - pos[d * 3 + 0];
  float ey = pos[s * 3 + 1] - pos[d * 3 + 1];
  float ez = pos[s * 3 + 2] - pos[d * 3 + 2];
  float r = sqrtf(ex * ex + ey * ey + ez * ez + 1e-12f);
  float inv = 1.f / r;
  float ux = ex * inv, uy = ey * inv, uz = ez * inv;
  const float s3 = 1.7320508075688772f;
  const float s15 = 3.872983346207417f;
  const float c20 = 1.118033988749895f;  // 0.5*sqrt(5)
  Y1g[e * 3 + 0] = s3 * ux;
  Y1g[e * 3 + 1] = s3 * uy;
  Y1g[e * 3 + 2] = s3 * uz;
  Y2g[e * 5 + 0] = s15 * ux * uy;
  Y2g[e * 5 + 1] = s15 * uy * uz;
  Y2g[e * 5 + 2] = c20 * (3.f * uz * uz - 1.f);
  Y2g[e * 5 + 3] = s15 * ux * uz;
  Y2g[e * 5 + 4] = 0.5f * s15 * (ux * ux - uy * uy);
  int bm = bond[e];
#pragma unroll
  for (int j = 0; j < 16; ++j) attr[(size_t)e * 32 + j] = (__bf16)ebond[bm * 16 + j];
  const float step = 5.0f / 17.0f;
#pragma unroll
  for (int j = 0; j < 16; ++j) {
    float t = (r - (float)(j + 1) * step) / step;
    attr[(size_t)e * 32 + 16 + j] = (__bf16)(__expf(-t * t) * 1.12f);
  }
}

// ---------------------------------------------------------------------------
// Kernel: x0 = atom_embed[z] @ W_init  (gathered-A WMMA, 16 nodes per tile)
// ---------------------------------------------------------------------------
__global__ void x0init_kernel(const float* __restrict__ atomE, const int* __restrict__ zI,
                              const float* __restrict__ Winit, float* __restrict__ x0g,
                              int numTiles) {
  __shared__ __bf16 W[32 * 64];  // col-major [col][K=64]
  int lane = threadIdx.x;
  for (int idx = lane; idx < 32 * 64; idx += 32) {
    int col = idx >> 6, k = idx & 63;
    W[col * 64 + k] = (__bf16)Winit[k * 32 + col];
  }
  __syncthreads();
  int row = lane & 15, hi = lane >> 4, col = lane & 15;
  for (int tile = blockIdx.x; tile < numTiles; tile += gridDim.x) {
    int nBase = tile * 16;
    const float* rp = atomE + (size_t)zI[nBase + row] * 64;
    v16bf A0 = packA_f32row(rp, 1, 0, 64);
    v16bf A1 = packA_f32row(rp, 1, 32, 64);
#pragma unroll
    for (int t = 0; t < 2; ++t) {
      v8f c = zero8();
      c = wmma_bf(A0, packB(W, 64, t * 16, 0, 16), c);
      c = wmma_bf(A1, packB(W, 64, t * 16, 32, 16), c);
#pragma unroll
      for (int v = 0; v < 8; ++v)
        x0g[(size_t)(nBase + v + 8 * hi) * 32 + t * 16 + col] = c[v];
    }
  }
}

// ---------------------------------------------------------------------------
// Kernel: q projections (per layer): q0=x0@Wq0, q1_i=x1_i@Wq1, q2_i=x2_i@Wq2
// ---------------------------------------------------------------------------
__global__ void qproj_kernel(const float* __restrict__ x0g, const float* __restrict__ x1g,
                             const float* __restrict__ x2g, const float* __restrict__ Wq0,
                             const float* __restrict__ Wq1, const float* __restrict__ Wq2,
                             float* __restrict__ q0g, float* __restrict__ q1g,
                             float* __restrict__ q2g, int numTiles) {
  __shared__ __bf16 W0[32 * 32];  // [col][K=32]
  __shared__ __bf16 W1[16 * 32];
  __shared__ __bf16 W2[8 * 32];
  int lane = threadIdx.x;
  for (int idx = lane; idx < 32 * 32; idx += 32) {
    int col = idx >> 5, k = idx & 31;
    W0[col * 32 + k] = (__bf16)Wq0[k * 32 + col];
  }
  for (int idx = lane; idx < 16 * 32; idx += 32) {
    int col = idx >> 5, k = idx & 31;
    W1[col * 32 + k] = (__bf16)(k < 16 ? Wq1[k * 16 + col] : 0.f);
  }
  for (int idx = lane; idx < 8 * 32; idx += 32) {
    int col = idx >> 5, k = idx & 31;
    W2[col * 32 + k] = (__bf16)(k < 8 ? Wq2[k * 8 + col] : 0.f);
  }
  __syncthreads();
  int row = lane & 15, hi = lane >> 4, col = lane & 15;
  for (int tile = blockIdx.x; tile < numTiles; tile += gridDim.x) {
    int nBase = tile * 16;
    {
      const float* rp = x0g + (size_t)(nBase + row) * 32;
      v16bf A = packA_f32row(rp, 1, 0, 32);
#pragma unroll
      for (int t = 0; t < 2; ++t) {
        v8f c = zero8();
        c = wmma_bf(A, packB(W0, 32, t * 16, 0, 16), c);
#pragma unroll
        for (int v = 0; v < 8; ++v)
          q0g[(size_t)(nBase + v + 8 * hi) * 32 + t * 16 + col] = c[v];
      }
    }
#pragma unroll
    for (int i = 0; i < 3; ++i) {
      const float* rp = x1g + (size_t)(nBase + row) * 48 + i;
      v16bf A = packA_f32row(rp, 3, 0, 16);
      v8f c = zero8();
      c = wmma_bf(A, packB(W1, 32, 0, 0, 16), c);
#pragma unroll
      for (int v = 0; v < 8; ++v)
        q1g[(size_t)(nBase + v + 8 * hi) * 48 + col * 3 + i] = c[v];
    }
#pragma unroll
    for (int i = 0; i < 5; ++i) {
      const float* rp = x2g + (size_t)(nBase + row) * 40 + i;
      v16bf A = packA_f32row(rp, 5, 0, 8);
      v8f c = zero8();
      c = wmma_bf(A, packB(W2, 32, 0, 0, 8), c);
      if (col < 8) {
#pragma unroll
        for (int v = 0; v < 8; ++v)
          q2g[(size_t)(nBase + v + 8 * hi) * 40 + col * 5 + i] = c[v];
      }
    }
  }
}

// ---------------------------------------------------------------------------
// Fused edge mega-kernel (per layer): edge MLP -> messages -> k/v WMMAs ->
// per-head logits + atomicMax segment max. One wave per 16-edge tile,
// grid-stride; weights staged once per block in LDS (col-major bf16).
// ---------------------------------------------------------------------------
struct KBuf {
  float k0[16 * 32];
  float k1[3 * 16 * 16];
  float k2[5 * 16 * 8];
};
union UScr {
  __bf16 hidden[16 * 64];
  KBuf k;
};
struct EdgeSmem {
  __bf16 We1[64 * 32];    // [col][K=32]
  __bf16 We2[144 * 64];   // [col][K=64]
  __bf16 Wk0[32 * 64];    // K padded 56->64
  __bf16 Wv0[32 * 64];
  __bf16 Wk1[16 * 64];    // K padded 48->64
  __bf16 Wv1[16 * 64];
  __bf16 Wk2[8 * 64];     // K padded 40->64
  __bf16 Wv2[8 * 64];
  float be1[64];
  float be2[144];
  __bf16 m0[16 * 64];      // pad cols 56..63 zero
  __bf16 m1[3 * 16 * 64];  // pad cols 48..63 zero
  __bf16 m2[5 * 16 * 64];  // pad cols 40..63 zero
  UScr u;
};  // 63808 bytes

__global__ void edge_attn_kernel(
    const __bf16* __restrict__ attr, const int* __restrict__ srcI,
    const int* __restrict__ dstI, const float* __restrict__ Y1g,
    const float* __restrict__ Y2g, const float* __restrict__ x0g,
    const float* __restrict__ x1g, const float* __restrict__ x2g,
    const float* __restrict__ q0g, const float* __restrict__ q1g,
    const float* __restrict__ q2g, const float* __restrict__ We1g,
    const float* __restrict__ be1g, const float* __restrict__ We2g,
    const float* __restrict__ be2g, const float* __restrict__ Wk0g,
    const float* __restrict__ Wk1g, const float* __restrict__ Wk2g,
    const float* __restrict__ Wv0g, const float* __restrict__ Wv1g,
    const float* __restrict__ Wv2g, float* __restrict__ v0g, float* __restrict__ v1g,
    float* __restrict__ v2g, float* __restrict__ logitg, int* __restrict__ mxg,
    int numTiles) {
  __shared__ EdgeSmem S;
  int lane = threadIdx.x;
  // ---- Stage weights (col-major bf16, zero-padded K) ----
  for (int idx = lane; idx < 64 * 32; idx += 32) {
    int col = idx >> 5, k = idx & 31;
    S.We1[col * 32 + k] = (__bf16)We1g[k * 64 + col];
  }
  for (int idx = lane; idx < 144 * 64; idx += 32) {
    int col = idx >> 6, k = idx & 63;
    S.We2[col * 64 + k] = (__bf16)We2g[k * 144 + col];
  }
  for (int idx = lane; idx < 32 * 64; idx += 32) {
    int col = idx >> 6, k = idx & 63;
    S.Wk0[col * 64 + k] = (__bf16)(k < 56 ? Wk0g[k * 32 + col] : 0.f);
    S.Wv0[col * 64 + k] = (__bf16)(k < 56 ? Wv0g[k * 32 + col] : 0.f);
  }
  for (int idx = lane; idx < 16 * 64; idx += 32) {
    int col = idx >> 6, k = idx & 63;
    S.Wk1[col * 64 + k] = (__bf16)(k < 48 ? Wk1g[k * 16 + col] : 0.f);
    S.Wv1[col * 64 + k] = (__bf16)(k < 48 ? Wv1g[k * 16 + col] : 0.f);
  }
  for (int idx = lane; idx < 8 * 64; idx += 32) {
    int col = idx >> 6, k = idx & 63;
    S.Wk2[col * 64 + k] = (__bf16)(k < 40 ? Wk2g[k * 8 + col] : 0.f);
    S.Wv2[col * 64 + k] = (__bf16)(k < 40 ? Wv2g[k * 8 + col] : 0.f);
  }
  for (int idx = lane; idx < 64; idx += 32) S.be1[idx] = be1g[idx];
  for (int idx = lane; idx < 144; idx += 32) S.be2[idx] = be2g[idx];
  // ---- Zero the constant K-padding columns of the message tiles ----
  for (int idx = lane; idx < 16 * 64; idx += 32)
    if ((idx & 63) >= 56) S.m0[idx] = (__bf16)0.f;
  for (int idx = lane; idx < 3 * 16 * 64; idx += 32)
    if ((idx & 63) >= 48) S.m1[idx] = (__bf16)0.f;
  for (int idx = lane; idx < 5 * 16 * 64; idx += 32)
    if ((idx & 63) >= 40) S.m2[idx] = (__bf16)0.f;
  __syncthreads();

  const float inv_scale = 0.25819888974716115f;  // 1/sqrt(15)
  int row = lane & 15, hi = lane >> 4, col = lane & 15;

  for (int tile = blockIdx.x; tile < numTiles; tile += gridDim.x) {
    int eBase = tile * 16;
    int nextTile = tile + gridDim.x;
    if (nextTile < numTiles)
      __builtin_prefetch(attr + (size_t)(nextTile * 16 + row) * 32, 0, 0);

    // ---- Stage 1: hidden = silu(attr @ We1 + be1), 16x64, 4 WMMAs ----
    v16bf Aattr = packA_bf16row(attr + (size_t)(eBase + row) * 32, hi);
#pragma unroll
    for (int t = 0; t < 4; ++t) {
      v8f c = splat8(S.be1[t * 16 + col]);
      c = wmma_bf(Aattr, packB(S.We1, 32, t * 16, 0, 16), c);
#pragma unroll
      for (int j = 0; j < 8; ++j) c[j] = silu(c[j]);
      storeC_ldsbf(S.u.hidden, 64, t * 16, c);
    }
    __syncthreads();

    // ---- Stage 2: w = hidden @ We2 + be2 (16x144) fused with message build ----
    v16bf Ah0 = packA_lds(S.u.hidden, 64, 0);
    v16bf Ah1 = packA_lds(S.u.hidden, 64, 32);
#pragma unroll
    for (int t = 0; t < 9; ++t) {
      v8f c = splat8(S.be2[t * 16 + col]);
      c = wmma_bf(Ah0, packB(S.We2, 64, t * 16, 0, 16), c);
      c = wmma_bf(Ah1, packB(S.We2, 64, t * 16, 32, 16), c);
      int cg = t * 16 + col;  // global w-channel handled by this lane
#pragma unroll
      for (int v = 0; v < 8; ++v) {
        int r = v + 8 * hi;
        int e = eBase + r;
        int s = srcI[e];
        float w = c[v];
        if (cg < 32) {  // ws0 * x0s
          S.m0[r * 64 + cg] = (__bf16)(w * x0g[(size_t)s * 32 + cg]);
        } else if (cg < 48) {  // ws1 * <x1s, Y1>
          int c1 = cg - 32;
          const float* xp = x1g + (size_t)s * 48 + c1 * 3;
          const float* y = Y1g + (size_t)e * 3;
          S.m0[r * 64 + cg] = (__bf16)(w * (xp[0] * y[0] + xp[1] * y[1] + xp[2] * y[2]));
        } else if (cg < 56) {  // ws2 * <x2s, Y2>
          int c2 = cg - 48;
          const float* xp = x2g + (size_t)s * 40 + c2 * 5;
          const float* y = Y2g + (size_t)e * 5;
          float dv = 0.f;
#pragma unroll
          for (int i = 0; i < 5; ++i) dv += xp[i] * y[i];
          S.m0[r * 64 + cg] = (__bf16)(w * dv);
        } else if (cg < 72) {  // wv1 * x1s  -> m1 channels [0,16)
          int cv = cg - 56;
          const float* xp = x1g + (size_t)s * 48 + cv * 3;
#pragma unroll
          for (int i = 0; i < 3; ++i)
            S.m1[i * 1024 + r * 64 + cv] = (__bf16)(w * xp[i]);
        } else if (cg < 104) {  // (wv0*x0s) x Y1 -> m1 channels [16,48)
          int cv = cg - 72;
          float t0 = w * x0g[(size_t)s * 32 + cv];
          const float* y = Y1g + (size_t)e * 3;
#pragma unroll
          for (int i = 0; i < 3; ++i)
            S.m1[i * 1024 + r * 64 + 16 + cv] = (__bf16)(t0 * y[i]);
        } else if (cg < 112) {  // wl2 * x2s -> m2 channels [0,8)
          int cl = cg - 104;
          const float* xp = x2g + (size_t)s * 40 + cl * 5;
#pragma unroll
          for (int i = 0; i < 5; ++i)
            S.m2[i * 1024 + r * 64 + cl] = (__bf16)(w * xp[i]);
        } else {  // (wl0*x0s) x Y2 -> m2 channels [8,40)
          int cl = cg - 112;
          float t0 = w * x0g[(size_t)s * 32 + cl];
          const float* y = Y2g + (size_t)e * 5;
#pragma unroll
          for (int i = 0; i < 5; ++i)
            S.m2[i * 1024 + r * 64 + 8 + cl] = (__bf16)(t0 * y[i]);
        }
      }
    }
    __syncthreads();

    // ---- Stage 3: k/v projections (WMMA, K padded to 64 = 2 k-steps) ----
    v16bf Am0a = packA_lds(S.m0, 64, 0);
    v16bf Am0b = packA_lds(S.m0, 64, 32);
#pragma unroll
    for (int t = 0; t < 2; ++t) {
      v8f c = zero8();
      c = wmma_bf(Am0a, packB(S.Wk0, 64, t * 16, 0, 16), c);
      c = wmma_bf(Am0b, packB(S.Wk0, 64, t * 16, 32, 16), c);
      storeC_ldsf(S.u.k.k0, 32, t * 16, c);
      v8f d = zero8();
      d = wmma_bf(Am0a, packB(S.Wv0, 64, t * 16, 0, 16), d);
      d = wmma_bf(Am0b, packB(S.Wv0, 64, t * 16, 32, 16), d);
#pragma unroll
      for (int v = 0; v < 8; ++v)
        v0g[(size_t)(eBase + v + 8 * hi) * 32 + t * 16 + col] = d[v];
    }
#pragma unroll
    for (int i = 0; i < 3; ++i) {
      v16bf Aa = packA_lds(S.m1 + i * 1024, 64, 0);
      v16bf Ab = packA_lds(S.m1 + i * 1024, 64, 32);
      v8f c = zero8();
      c = wmma_bf(Aa, packB(S.Wk1, 64, 0, 0, 16), c);
      c = wmma_bf(Ab, packB(S.Wk1, 64, 0, 32, 16), c);
      storeC_ldsf(S.u.k.k1 + i * 256, 16, 0, c);
      v8f d = zero8();
      d = wmma_bf(Aa, packB(S.Wv1, 64, 0, 0, 16), d);
      d = wmma_bf(Ab, packB(S.Wv1, 64, 0, 32, 16), d);
#pragma unroll
      for (int v = 0; v < 8; ++v)
        v1g[(size_t)(eBase + v + 8 * hi) * 48 + col * 3 + i] = d[v];
    }
#pragma unroll
    for (int i = 0; i < 5; ++i) {
      v16bf Aa = packA_lds(S.m2 + i * 1024, 64, 0);
      v16bf Ab = packA_lds(S.m2 + i * 1024, 64, 32);
      v8f c = zero8();
      c = wmma_bf(Aa, packB(S.Wk2, 64, 0, 0, 8), c);
      c = wmma_bf(Ab, packB(S.Wk2, 64, 0, 32, 8), c);
      if (col < 8) {
#pragma unroll
        for (int v = 0; v < 8; ++v)
          S.u.k.k2[i * 128 + (v + 8 * hi) * 8 + col] = c[v];
      }
      v8f d = zero8();
      d = wmma_bf(Aa, packB(S.Wv2, 64, 0, 0, 8), d);
      d = wmma_bf(Ab, packB(S.Wv2, 64, 0, 32, 8), d);
      if (col < 8) {
#pragma unroll
        for (int v = 0; v < 8; ++v)
          v2g[(size_t)(eBase + v + 8 * hi) * 40 + col * 5 + i] = d[v];
      }
    }
    __syncthreads();

    // ---- Stage 4: per-head logits + segment atomicMax ----
    {
      int e = eBase + row;  // lane pairs (row, hi) split the 8 heads
      int dn = dstI[e];
      const float* q0p = q0g + (size_t)dn * 32;
      const float* q1p = q1g + (size_t)dn * 48;
      const float* q2p = q2g + (size_t)dn * 40;
#pragma unroll
      for (int hh = 0; hh < 4; ++hh) {
        int h = hi * 4 + hh;
        float s = 0.f;
#pragma unroll
        for (int j = 0; j < 4; ++j)
          s += q0p[h * 4 + j] * S.u.k.k0[row * 32 + h * 4 + j];
#pragma unroll
        for (int dd = 0; dd < 2; ++dd) {
          int d1 = h * 2 + dd;
#pragma unroll
          for (int i = 0; i < 3; ++i)
            s += q1p[d1 * 3 + i] * S.u.k.k1[i * 256 + row * 16 + d1];
        }
#pragma unroll
        for (int i = 0; i < 5; ++i)
          s += q2p[h * 5 + i] * S.u.k.k2[i * 128 + row * 8 + h];
        s *= inv_scale;
        logitg[(size_t)e * 8 + h] = s;
        atomicMax(mxg + (size_t)dn * 8 + h, fenc(s));
      }
    }
    __syncthreads();
  }
}

// ---------------------------------------------------------------------------
// Softmax passes and scatter
// ---------------------------------------------------------------------------
__global__ void initmx_kernel(int* mx, int n) {
  int i = blockIdx.x * blockDim.x + threadIdx.x;
  if (i < n) mx[i] = (int)0x80000000;
}

__global__ void softmax_exp_kernel(const float* __restrict__ logitg,
                                   const int* __restrict__ mxg,
                                   const int* __restrict__ dstI,
                                   float* __restrict__ aexp, float* __restrict__ den,
                                   int total) {
  int i = blockIdx.x * blockDim.x + threadIdx.x;
  if (i >= total) return;
  int e = i >> 3, h = i & 7;
  int dn = dstI[e];
  float a = __expf(logitg[i] - fdec(mxg[dn * 8 + h]));
  aexp[i] = a;
  atomicAdd(den + dn * 8 + h, a);
}

__global__ void scatter_kernel(const float* __restrict__ aexp,
                               const float* __restrict__ den,
                               const int* __restrict__ dstI,
                               const float* __restrict__ v0g,
                               const float* __restrict__ v1g,
                               const float* __restrict__ v2g, float* __restrict__ acc0,
                               float* __restrict__ acc1, float* __restrict__ acc2,
                               int E) {
  int e = blockIdx.x * blockDim.x + threadIdx.x;
  if (e >= E) return;
  int dn = dstI[e];
  float al[8];
#pragma unroll
  for (int h = 0; h < 8; ++h)
    al[h] = aexp[(size_t)e * 8 + h] / (den[dn * 8 + h] + 1e-9f);
#pragma unroll
  for (int c = 0; c < 32; ++c)
    atomicAdd(acc0 + (size_t)dn * 32 + c, al[c >> 2] * v0g[(size_t)e * 32 + c]);
#pragma unroll
  for (int c = 0; c < 16; ++c)
#pragma unroll
    for (int i = 0; i < 3; ++i)
      atomicAdd(acc1 + (size_t)dn * 48 + c * 3 + i,
                al[c >> 1] * v1g[(size_t)e * 48 + c * 3 + i]);
#pragma unroll
  for (int c = 0; c < 8; ++c)
#pragma unroll
    for (int i = 0; i < 5; ++i)
      atomicAdd(acc2 + (size_t)dn * 40 + c * 5 + i,
                al[c] * v2g[(size_t)e * 40 + c * 5 + i]);
}

// ---------------------------------------------------------------------------
// Node update: x += acc @ W_o (WMMA per 16-node tile)
// ---------------------------------------------------------------------------
__global__ void nodeupd_kernel(const float* __restrict__ acc0,
                               const float* __restrict__ acc1,
                               const float* __restrict__ acc2,
                               const float* __restrict__ Wo0,
                               const float* __restrict__ Wo1,
                               const float* __restrict__ Wo2, float* __restrict__ x0g,
                               float* __restrict__ x1g, float* __restrict__ x2g,
                               int numTiles) {
  __shared__ __bf16 W0[32 * 32];
  __shared__ __bf16 W1[16 * 32];
  __shared__ __bf16 W2[8 * 32];
  int lane = threadIdx.x;
  for (int idx = lane; idx < 32 * 32; idx += 32) {
    int col = idx >> 5, k = idx & 31;
    W0[col * 32 + k] = (__bf16)Wo0[k * 32 + col];
  }
  for (int idx = lane; idx < 16 * 32; idx += 32) {
    int col = idx >> 5, k = idx & 31;
    W1[col * 32 + k] = (__bf16)(k < 16 ? Wo1[k * 16 + col] : 0.f);
  }
  for (int idx = lane; idx < 8 * 32; idx += 32) {
    int col = idx >> 5, k = idx & 31;
    W2[col * 32 + k] = (__bf16)(k < 8 ? Wo2[k * 8 + col] : 0.f);
  }
  __syncthreads();
  int row = lane & 15, hi = lane >> 4, col = lane & 15;
  for (int tile = blockIdx.x; tile < numTiles; tile += gridDim.x) {
    int nBase = tile * 16;
    {
      const float* rp = acc0 + (size_t)(nBase + row) * 32;
      v16bf A = packA_f32row(rp, 1, 0, 32);
#pragma unroll
      for (int t = 0; t < 2; ++t) {
        v8f c = zero8();
        c = wmma_bf(A, packB(W0, 32, t * 16, 0, 16), c);
#pragma unroll
        for (int v = 0; v < 8; ++v)
          x0g[(size_t)(nBase + v + 8 * hi) * 32 + t * 16 + col] += c[v];
      }
    }
#pragma unroll
    for (int i = 0; i < 3; ++i) {
      const float* rp = acc1 + (size_t)(nBase + row) * 48 + i;
      v16bf A = packA_f32row(rp, 3, 0, 16);
      v8f c = zero8();
      c = wmma_bf(A, packB(W1, 32, 0, 0, 16), c);
#pragma unroll
      for (int v = 0; v < 8; ++v)
        x1g[(size_t)(nBase + v + 8 * hi) * 48 + col * 3 + i] += c[v];
    }
#pragma unroll
    for (int i = 0; i < 5; ++i) {
      const float* rp = acc2 + (size_t)(nBase + row) * 40 + i;
      v16bf A = packA_f32row(rp, 5, 0, 8);
      v8f c = zero8();
      c = wmma_bf(A, packB(W2, 32, 0, 0, 8), c);
      if (col < 8) {
#pragma unroll
        for (int v = 0; v < 8; ++v)
          x2g[(size_t)(nBase + v + 8 * hi) * 40 + col * 5 + i] += c[v];
      }
    }
  }
}

// ---------------------------------------------------------------------------
// Graph pooling + invariant head
// ---------------------------------------------------------------------------
__global__ void pool_kernel(const float* __restrict__ x0g, const float* __restrict__ x1g,
                            const float* __restrict__ x2g, const int* __restrict__ batch,
                            float* __restrict__ g0, float* __restrict__ g1,
                            float* __restrict__ g2, int N) {
  int n = blockIdx.x * blockDim.x + threadIdx.x;
  if (n >= N) return;
  int b = batch[n];
#pragma unroll
  for (int c = 0; c < 32; ++c) atomicAdd(g0 + b * 32 + c, x0g[(size_t)n * 32 + c]);
#pragma unroll
  for (int j = 0; j < 48; ++j) atomicAdd(g1 + b * 48 + j, x1g[(size_t)n * 48 + j]);
#pragma unroll
  for (int j = 0; j < 40; ++j) atomicAdd(g2 + b * 40 + j, x2g[(size_t)n * 40 + j]);
}

__global__ void head_kernel(const float* __restrict__ g0, const float* __restrict__ g1,
                            const float* __restrict__ g2, const float* __restrict__ Wh1,
                            const float* __restrict__ bh1, const float* __restrict__ Wh2,
                            float* __restrict__ out, int NG) {
  int g = threadIdx.x;
  if (g >= NG) return;
  float inv[56];
#pragma unroll
  for (int c = 0; c < 32; ++c) inv[c] = g0[g * 32 + c];
#pragma unroll
  for (int c = 0; c < 16; ++c) {
    float s = 1e-12f;
#pragma unroll
    for (int i = 0; i < 3; ++i) {
      float t = g1[g * 48 + c * 3 + i];
      s += t * t;
    }
    inv[32 + c] = sqrtf(s);
  }
#pragma unroll
  for (int c = 0; c < 8; ++c) {
    float s = 1e-12f;
#pragma unroll
    for (int i = 0; i < 5; ++i) {
      float t = g2[g * 40 + c * 5 + i];
      s += t * t;
    }
    inv[48 + c] = sqrtf(s);
  }
  float o = 0.f;
  for (int j = 0; j < 56; ++j) {
    float s = bh1[j];
    for (int k = 0; k < 56; ++k) s += inv[k] * Wh1[k * 56 + j];
    o += silu(s) * Wh2[j];
  }
  out[g] = o;
}

// ---------------------------------------------------------------------------
// Host launcher
// ---------------------------------------------------------------------------
extern "C" void kernel_launch(void* const* d_in, const int* in_sizes, int n_in,
                              void* d_out, int out_size, void* d_ws, size_t ws_size,
                              hipStream_t stream) {
  const int E = E_EDGES, N = N_NODES_C, NG = N_GRAPHS_C;
  const float* pos = (const float*)d_in[0];
  const int* ei = (const int*)d_in[1];
  const int* srcI = ei;
  const int* dstI = ei + E;
  const int* bond = (const int*)d_in[2];
  const int* zI = (const int*)d_in[3];
  const int* batch = (const int*)d_in[4];
  const float* ebond = (const float*)d_in[6];
  const float* atomE = (const float*)d_in[7];
  const float* Winit = (const float*)d_in[8];
  const float* We1 = (const float*)d_in[9];
  const float* be1 = (const float*)d_in[10];
  const float* We2 = (const float*)d_in[11];
  const float* be2 = (const float*)d_in[12];
  const float* Wq0 = (const float*)d_in[13];
  const float* Wq1 = (const float*)d_in[14];
  const float* Wq2 = (const float*)d_in[15];
  const float* Wk0 = (const float*)d_in[16];
  const float* Wk1 = (const float*)d_in[17];
  const float* Wk2 = (const float*)d_in[18];
  const float* Wv0 = (const float*)d_in[19];
  const float* Wv1 = (const float*)d_in[20];
  const float* Wv2 = (const float*)d_in[21];
  const float* Wo0 = (const float*)d_in[22];
  const float* Wo1 = (const float*)d_in[23];
  const float* Wo2 = (const float*)d_in[24];
  const float* Wh1 = (const float*)d_in[25];
  const float* bh1 = (const float*)d_in[26];
  const float* Wh2 = (const float*)d_in[27];

  unsigned char* ws = (unsigned char*)d_ws;
  size_t off = 0;
  auto alloc = [&](size_t bytes) -> void* {
    void* p = ws + off;
    off = (off + bytes + 255) & ~(size_t)255;
    return p;
  };
  __bf16* attr = (__bf16*)alloc((size_t)E * 32 * 2);
  float* Y1g = (float*)alloc((size_t)E * 3 * 4);
  float* Y2g = (float*)alloc((size_t)E * 5 * 4);
  float* x0g = (float*)alloc((size_t)N * 32 * 4);
  float* x1g = (float*)alloc((size_t)N * 48 * 4);
  float* x2g = (float*)alloc((size_t)N * 40 * 4);
  float* q0g = (float*)alloc((size_t)N * 32 * 4);
  float* q1g = (float*)alloc((size_t)N * 48 * 4);
  float* q2g = (float*)alloc((size_t)N * 40 * 4);
  float* v0g = (float*)alloc((size_t)E * 32 * 4);
  float* v1g = (float*)alloc((size_t)E * 48 * 4);
  float* v2g = (float*)alloc((size_t)E * 40 * 4);
  float* logitg = (float*)alloc((size_t)E * 8 * 4);
  float* aexp = (float*)alloc((size_t)E * 8 * 4);
  int* mxg = (int*)alloc((size_t)N * 8 * 4);
  float* den = (float*)alloc((size_t)N * 8 * 4);
  float* acc0 = (float*)alloc((size_t)N * 32 * 4);
  float* acc1 = (float*)alloc((size_t)N * 48 * 4);
  float* acc2 = (float*)alloc((size_t)N * 40 * 4);
  float* g0 = (float*)alloc((size_t)NG * 32 * 4);
  float* g1 = (float*)alloc((size_t)NG * 48 * 4);
  float* g2 = (float*)alloc((size_t)NG * 40 * 4);
  (void)ws_size;
  (void)in_sizes;
  (void)n_in;
  (void)out_size;

  hipMemsetAsync(x1g, 0, (size_t)N * 48 * 4, stream);
  hipMemsetAsync(x2g, 0, (size_t)N * 40 * 4, stream);
  geom_kernel<<<(E + 255) / 256, 256, 0, stream>>>(pos, srcI, dstI, bond, ebond, attr,
                                                   Y1g, Y2g, E);
  x0init_kernel<<<625, 32, 0, stream>>>(atomE, zI, Winit, x0g, 625);

  for (int l = 0; l < 2; ++l) {
    initmx_kernel<<<(N * 8 + 255) / 256, 256, 0, stream>>>(mxg, N * 8);
    hipMemsetAsync(den, 0, (size_t)N * 8 * 4, stream);
    hipMemsetAsync(acc0, 0, (size_t)N * 32 * 4, stream);
    hipMemsetAsync(acc1, 0, (size_t)N * 48 * 4, stream);
    hipMemsetAsync(acc2, 0, (size_t)N * 40 * 4, stream);
    qproj_kernel<<<625, 32, 0, stream>>>(x0g, x1g, x2g, Wq0 + l * 32 * 32,
                                         Wq1 + l * 16 * 16, Wq2 + l * 8 * 8, q0g, q1g,
                                         q2g, 625);
    edge_attn_kernel<<<1280, 32, 0, stream>>>(
        attr, srcI, dstI, Y1g, Y2g, x0g, x1g, x2g, q0g, q1g, q2g, We1 + l * 32 * 64,
        be1 + l * 64, We2 + l * 64 * 144, be2 + l * 144, Wk0 + l * 56 * 32,
        Wk1 + l * 48 * 16, Wk2 + l * 40 * 8, Wv0 + l * 56 * 32, Wv1 + l * 48 * 16,
        Wv2 + l * 40 * 8, v0g, v1g, v2g, logitg, mxg, E / 16);
    softmax_exp_kernel<<<(E * 8 + 255) / 256, 256, 0, stream>>>(logitg, mxg, dstI, aexp,
                                                                den, E * 8);
    scatter_kernel<<<(E + 255) / 256, 256, 0, stream>>>(aexp, den, dstI, v0g, v1g, v2g,
                                                        acc0, acc1, acc2, E);
    nodeupd_kernel<<<625, 32, 0, stream>>>(acc0, acc1, acc2, Wo0 + l * 32 * 32,
                                           Wo1 + l * 16 * 16, Wo2 + l * 8 * 8, x0g, x1g,
                                           x2g, 625);
  }
  hipMemsetAsync(g0, 0, (size_t)NG * 32 * 4, stream);
  hipMemsetAsync(g1, 0, (size_t)NG * 48 * 4, stream);
  hipMemsetAsync(g2, 0, (size_t)NG * 40 * 4, stream);
  pool_kernel<<<(N + 255) / 256, 256, 0, stream>>>(x0g, x1g, x2g, batch, g0, g1, g2, N);
  head_kernel<<<1, 64, 0, stream>>>(g0, g1, g2, Wh1, bh1, Wh2, (float*)d_out, NG);
}
